// DTransformer_51084341019395
// MI455X (gfx1250) — matrix operand order repaired and checked by reading, not compiled, not run on hardware
//
#include <hip/hip_runtime.h>
#include <hip/hip_bf16.h>

#define BS   8
#define SEQ  1024
#define DM   256
#define NH   8
#define DK   32
#define NEGV (-1.0e32f)

typedef __attribute__((ext_vector_type(16))) _Float16 v16h;
typedef __attribute__((ext_vector_type(8)))  float    v8f;
typedef __attribute__((ext_vector_type(4))) unsigned int u32x4;
typedef __attribute__((ext_vector_type(8))) int          i32x8;
typedef __attribute__((ext_vector_type(4))) int          i32x4;

__device__ __forceinline__ float waveReduceMax(float v) {
#pragma unroll
  for (int o = 16; o > 0; o >>= 1) v = fmaxf(v, __shfl_xor(v, o, 32));
  return v;
}
__device__ __forceinline__ float waveReduceSum(float v) {
#pragma unroll
  for (int o = 16; o > 0; o >>= 1) v += __shfl_xor(v, o, 32);
  return v;
}

// ---------------------------------------------------------------------------
// Tensor Data Mover: DMA a 2D f32 tile (tileW x tileH elements, row stride
// strideElems) from global memory into LDS.  Descriptor per CDNA5 ISA §8.3/8.4:
//   group0: count=1 | lds_addr | global_addr[56:0] | type=2
//   group1: data_size=4B, tensor dims, tile dims, dim0 stride
// Issue once per workgroup (TDM ignores EXEC); caller waits TENSORcnt.
// ---------------------------------------------------------------------------
__device__ __forceinline__ void tdm_load_tile_f32(
    void* ldsDst, const float* src, unsigned tileW, unsigned tileH,
    unsigned tensorW, unsigned tensorH, unsigned long long strideElems) {
  unsigned lds = (unsigned)(size_t)ldsDst;              // generic low 32b = LDS byte offset
  unsigned long long ga = (unsigned long long)(size_t)src;
  u32x4 g0;
  g0[0] = 1u;                                           // count=1, user mode
  g0[1] = lds;                                          // lds_addr (bytes)
  g0[2] = (unsigned)(ga & 0xffffffffu);                 // global_addr[31:0]
  g0[3] = (unsigned)((ga >> 32) & 0x01ffffffu)          // global_addr[56:32]
        | (2u << 30);                                   // type = 2 ("image")
  i32x8 g1;
  g1[0] = (int)(2u << 16);                              // wg_mask=0, data_size=2 (4B)
  g1[1] = (int)((tensorW & 0xffffu) << 16);             // tensor_dim0[15:0] @ bits 63:48
  g1[2] = (int)((tensorW >> 16) | ((tensorH & 0xffffu) << 16)); // dim0 hi | dim1 lo
  g1[3] = (int)((tensorH >> 16) | (tileW << 16));       // dim1 hi | tile_dim0
  g1[4] = (int)(tileH & 0xffffu);                       // tile_dim1 (tile_dim2=0)
  g1[5] = (int)(strideElems & 0xffffffffu);             // tensor_dim0_stride[31:0]
  g1[6] = (int)((strideElems >> 32) & 0xffffu);         // stride0[47:32], stride1 lo = 0
  g1[7] = 0;                                            // stride1 hi = 0
  i32x4 z4 = {0, 0, 0, 0};
#if __clang_major__ >= 23
  i32x8 z8 = {0, 0, 0, 0, 0, 0, 0, 0};
  __builtin_amdgcn_tensor_load_to_lds(g0, g1, z4, z4, z8, 0);
#else
  __builtin_amdgcn_tensor_load_to_lds(g0, g1, z4, z4, 0);
#endif
}

// ---------------------------------------------------------------------------
// GEMM: Y[R,256] = X[R,256] @ W[256,256]^T + bias.  One wave per 16x16 tile,
// 8 waves/block cover 16 rows x 128 cols; X tile staged into LDS by the TDM.
// ---------------------------------------------------------------------------
__global__ void __launch_bounds__(256) gemm256_wt_bias(
    const float* __restrict__ X, const float* __restrict__ W,
    const float* __restrict__ bias, float* __restrict__ Y) {
  __shared__ float sX[16 * 256];
  const int wave = threadIdx.x >> 5;
  const int lane = threadIdx.x & 31;
  const int rowBase = blockIdx.x * 16;
  const int colBase = blockIdx.y * 128 + wave * 16;

  const int m  = lane & 15;   // A-row / B-col / C-col index for this lane
  const int hi = lane >> 4;

  // warm L2 for this wave's 16 weight rows while the TDM fills LDS
  __builtin_prefetch(W + (size_t)(colBase + m) * 256, 0, 1);

  if (wave == 0) {
    tdm_load_tile_f32(sX, X + (size_t)rowBase * 256,
                      /*tileW=*/256, /*tileH=*/16,
                      /*tensorW=*/256, /*tensorH=*/16, /*stride=*/256);
    __builtin_amdgcn_s_wait_tensorcnt(0);
  }
  __syncthreads();

  v8f acc = {};
#pragma unroll
  for (int kc = 0; kc < 8; ++kc) {
    const int kb = kc * 32;
    v16h a, b;
#pragma unroll
    for (int v = 0; v < 8; ++v) {
      int ka = 2 * v + ((v >= 4) ? 8 : 0) + hi * 8;   // A frag K mapping
      a[2 * v]     = (_Float16)sX[m * 256 + kb + ka];
      a[2 * v + 1] = (_Float16)sX[m * 256 + kb + ka + 1];
      int kbv = 2 * v + ((v >= 4) ? 8 : 0) + hi * 8;   // B frag K mapping
      const float* wp = W + (size_t)(colBase + m) * 256 + kb + kbv;
      b[2 * v]     = (_Float16)wp[0];
      b[2 * v + 1] = (_Float16)wp[1];
    }
    acc = __builtin_amdgcn_wmma_f32_16x16x32_f16(false, a, false, b,
                                                 (short)0, acc, false, false);
  }
#pragma unroll
  for (int v = 0; v < 8; ++v) {
    int cm  = v + hi * 8;
    int col = colBase + m;
    Y[(size_t)(rowBase + cm) * 256 + col] = acc[v] + bias[col];
  }
}

// ---------------------------------------------------------------------------
// Fused distance-decay attention. One block per (b, h, 16-row q-tile).
// LDS: scores [16][1024] f32 (64KB) + probs [16][1024] f16 (32KB), dynamic.
// ---------------------------------------------------------------------------
__global__ void __launch_bounds__(256) attn_decay_kernel(
    const float* __restrict__ Qb, int qBatchStride, int qRowStride,
    const float* __restrict__ Km, const float* __restrict__ Vm,
    const float* __restrict__ g, float* __restrict__ ctx, int peek) {
  extern __shared__ char smemRaw[];
  float*    sS = (float*)smemRaw;                       // 16*1024 f32
  _Float16* sP = (_Float16*)(smemRaw + 16 * 1024 * 4);  // 16*1024 f16

  const int wave  = threadIdx.x >> 5;
  const int lane  = threadIdx.x & 31;
  const int qtile = blockIdx.x * 16;
  const int h     = blockIdx.y;
  const int b     = blockIdx.z;
  const int hOff  = h * DK;

  const int m  = lane & 15;
  const int hi = lane >> 4;

  // ---- Q fragment (reused by all 8 score tiles of this wave) ----
  v16h aq;
#pragma unroll
  for (int v = 0; v < 8; ++v) {
    int ka = 2 * v + ((v >= 4) ? 8 : 0) + hi * 8;
    size_t qi = (size_t)b * qBatchStride + (size_t)(qtile + m) * qRowStride + hOff;
    aq[2 * v]     = (_Float16)Qb[qi + ka];
    aq[2 * v + 1] = (_Float16)Qb[qi + ka + 1];
  }

  // ---- scores[16][1024] = Q_tile @ K^T / sqrt(dk); 8 key tiles per wave ----
#pragma unroll
  for (int t = 0; t < 8; ++t) {
    const int keyBase = (wave * 8 + t) * 16;
    v16h bk;
#pragma unroll
    for (int v = 0; v < 8; ++v) {
      int kbv = 2 * v + ((v >= 4) ? 8 : 0) + hi * 8;
      const float* kp = Km + ((size_t)b * SEQ + keyBase + m) * 256 + hOff + kbv;
      bk[2 * v]     = (_Float16)kp[0];
      bk[2 * v + 1] = (_Float16)kp[1];
    }
    v8f acc = {};
    acc = __builtin_amdgcn_wmma_f32_16x16x32_f16(false, aq, false, bk,
                                                 (short)0, acc, false, false);
#pragma unroll
    for (int v = 0; v < 8; ++v) {
      int cm = v + hi * 8;
      sS[cm * 1024 + keyBase + m] = acc[v] * 0.17677669529663687f; // 1/sqrt(32)
    }
  }
  __syncthreads();

  // ---- masked softmax -> cumsum -> distance decay -> second softmax ----
  const float gamma = -log1pf(expf(g[h]));   // -softplus(g_h)
  for (int rr = 0; rr < 2; ++rr) {
    const int r    = wave * 2 + rr;
    const int qpos = qtile + r;
    const int lim  = qpos - (peek ? 0 : 1);  // key j valid iff j <= lim
    float* row = sS + r * 1024;
    const int j0 = lane * 32;

    float mx = -3.0e38f;
    for (int jj = 0; jj < 32; ++jj) {
      int j = j0 + jj;
      mx = fmaxf(mx, (j <= lim) ? row[j] : NEGV);
    }
    mx = waveReduceMax(mx);

    float sm = 0.f;
    for (int jj = 0; jj < 32; ++jj) {
      int j = j0 + jj;
      sm += expf(((j <= lim) ? row[j] : NEGV) - mx);
    }
    sm = waveReduceSum(sm);
    const float inv = 1.0f / sm;

    // lane-local probability mass + exclusive prefix across the wave
    float lp = 0.f;
    for (int jj = 0; jj < 32; ++jj) {
      int j = j0 + jj;
      lp += expf(((j <= lim) ? row[j] : NEGV) - mx) * inv;
    }
    float pre = lp;
#pragma unroll
    for (int o = 1; o < 32; o <<= 1) {
      float t2 = __shfl_up(pre, o, 32);
      if (lane >= o) pre += t2;
    }
    float run = pre - lp;   // exclusive prefix (disttot == 1 by construction)

    float mx2 = -3.0e38f;
    for (int jj = 0; jj < 32; ++jj) {
      int j = j0 + jj;
      float sRaw = row[j];
      float p = expf(((j <= lim) ? sRaw : NEGV) - mx) * inv;
      run += p;                                   // inclusive cumsum
      float tail = fmaxf(1.0f - run, 0.0f);       // disttot - distcum
      float pos  = fabsf((float)(qpos - j));
      float dist = sqrtf(tail * pos);
      float eff  = fminf(fmaxf(expf(dist * gamma), 1e-5f), 1e5f);
      float s2   = (j <= lim) ? sRaw * eff : NEGV;
      row[j] = s2;
      mx2 = fmaxf(mx2, s2);
    }
    mx2 = waveReduceMax(mx2);

    float sm2 = 0.f;
    for (int jj = 0; jj < 32; ++jj) sm2 += expf(row[j0 + jj] - mx2);
    sm2 = waveReduceSum(sm2);
    const float inv2 = 1.0f / sm2;

    for (int jj = 0; jj < 32; ++jj) {
      int j = j0 + jj;
      float a = (j <= lim) ? expf(row[j] - mx2) * inv2 : 0.0f;
      sP[r * 1024 + j] = (_Float16)a;
    }
  }
  __syncthreads();

  // ---- out_tile[16,32] = A[16,1024] @ V[1024,32]; 4 key-chunks per wave ----
  v8f acc0 = {}, acc1 = {};
  for (int c = wave; c < 32; c += 8) {
    const int kb = c * 32;
    v16h ap;
#pragma unroll
    for (int v = 0; v < 8; ++v) {
      int ka = 2 * v + ((v >= 4) ? 8 : 0) + hi * 8;
      ap[2 * v]     = sP[m * 1024 + kb + ka];
      ap[2 * v + 1] = sP[m * 1024 + kb + ka + 1];
    }
#pragma unroll
    for (int nt = 0; nt < 2; ++nt) {
      v16h bv;
#pragma unroll
      for (int v = 0; v < 8; ++v) {
        int kv = 2 * v + ((v >= 4) ? 8 : 0) + hi * 8;
        const float* vp = Vm + ((size_t)b * SEQ + kb + kv) * 256 + hOff + nt * 16 + m;
        bv[2 * v]     = (_Float16)vp[0];
        bv[2 * v + 1] = (_Float16)vp[256];   // next key row
      }
      v8f& acc = nt ? acc1 : acc0;
      acc = __builtin_amdgcn_wmma_f32_16x16x32_f16(false, ap, false, bv,
                                                   (short)0, acc, false, false);
    }
  }

  // reduce 8 wave-partials via LDS (reuse score storage)
  float* sPart = sS;
#pragma unroll
  for (int v = 0; v < 8; ++v) {
    int cm = v + hi * 8;
    sPart[(wave * 16 + cm) * 32 + m]      = acc0[v];
    sPart[(wave * 16 + cm) * 32 + 16 + m] = acc1[v];
  }
  __syncthreads();
  for (int idx = threadIdx.x; idx < 512; idx += 256) {
    int om = idx >> 5, on = idx & 31;
    float s = 0.f;
#pragma unroll
    for (int w = 0; w < 8; ++w) s += sPart[(w * 16 + om) * 32 + on];
    ctx[((size_t)b * SEQ + qtile + om) * 256 + hOff + on] = s;
  }
}

// ---------------------------------------------------------------------------
// x = resid + y; LayerNorm(x) * lng + lnb.  residRowStride==0 broadcasts.
// ---------------------------------------------------------------------------
__global__ void __launch_bounds__(256) resln_kernel(
    const float* __restrict__ resid, int residRowStride,
    const float* __restrict__ y, const float* __restrict__ lng,
    const float* __restrict__ lnb, float* __restrict__ out) {
  const int wave = threadIdx.x >> 5;
  const int lane = threadIdx.x & 31;
  const size_t row = (size_t)blockIdx.x * 8 + wave;
  float xs[8];
  float sum = 0.f, sq = 0.f;
#pragma unroll
  for (int i = 0; i < 8; ++i) {
    int idx = lane + i * 32;
    float x = resid[row * (size_t)residRowStride + idx] + y[row * 256 + idx];
    xs[i] = x; sum += x; sq += x * x;
  }
  sum = waveReduceSum(sum);
  sq  = waveReduceSum(sq);
  float mean = sum * (1.0f / 256.0f);
  float var  = sq * (1.0f / 256.0f) - mean * mean;
  float rstd = rsqrtf(fmaxf(var, 0.0f) + 1e-5f);
#pragma unroll
  for (int i = 0; i < 8; ++i) {
    int idx = lane + i * 32;
    out[row * 256 + idx] = (xs[i] - mean) * rstd * lng[idx] + lnb[idx];
  }
}

// ---------------------------------------------------------------------------
// q3 = know @ Wq3^T + bq3   and   kk = sigmoid(know_heads @ Wlk^T + blk)
// ---------------------------------------------------------------------------
__global__ void __launch_bounds__(256) know_proj_kernel(
    const float* __restrict__ know, const float* __restrict__ Wq3,
    const float* __restrict__ bq3, const float* __restrict__ Wlk,
    const float* __restrict__ blk, float* __restrict__ q3,
    float* __restrict__ kk) {
  __shared__ float sk[256];
  const int t = threadIdx.x;
  sk[t] = know[t];
  __syncthreads();
  float acc = bq3[t];
  for (int k = 0; k < 256; ++k) acc += sk[k] * Wq3[(size_t)t * 256 + k];
  q3[t] = acc;
  for (int h = 0; h < NH; ++h) {
    float a = blk[t];
#pragma unroll
    for (int k = 0; k < 32; ++k) a += sk[h * 32 + k] * Wlk[(size_t)t * 32 + k];
    kk[h * 256 + t] = 1.0f / (1.0f + expf(-a));
  }
}

// ---------------------------------------------------------------------------
// Final stage: beta -> alpha (softmax over heads) -> sum_h alpha*sigmoid(Wlv..)
// One block per (b, s).
// ---------------------------------------------------------------------------
__global__ void __launch_bounds__(256) final_mix_kernel(
    const float* __restrict__ q_emb, const float* __restrict__ hh,
    const float* __restrict__ kk, const float* __restrict__ Wlv,
    const float* __restrict__ blv, float* __restrict__ out) {
  __shared__ float sQ[256], sH[256], sBeta[8];
  const int t = threadIdx.x;
  const int wave = t >> 5, lane = t & 31;
  const size_t row = (size_t)blockIdx.z * SEQ + blockIdx.x;
  sQ[t] = q_emb[row * 256 + t];
  sH[t] = hh[row * 256 + t];
  __syncthreads();
  float part = 0.f;
#pragma unroll
  for (int i = 0; i < 8; ++i) {
    int idx = lane + i * 32;
    part += kk[wave * 256 + idx] * sQ[idx];
  }
  part = waveReduceSum(part);
  if (lane == 0) sBeta[wave] = part;
  __syncthreads();
  float bmax = -3.0e38f;
#pragma unroll
  for (int h = 0; h < 8; ++h) bmax = fmaxf(bmax, sBeta[h]);
  float den = 0.f, eh[8];
#pragma unroll
  for (int h = 0; h < 8; ++h) { eh[h] = expf(sBeta[h] - bmax); den += eh[h]; }
  const float rden = 1.0f / den;
  float acc = 0.f;
#pragma unroll
  for (int h = 0; h < 8; ++h) {
    float d = blv[t];
#pragma unroll
    for (int k = 0; k < 32; ++k) d += sH[h * 32 + k] * Wlv[(size_t)t * 32 + k];
    acc += eh[h] * rden * (1.0f / (1.0f + expf(-d)));
  }
  out[row * 256 + t] = acc;
}

// ---------------------------------------------------------------------------
extern "C" void kernel_launch(void* const* d_in, const int* in_sizes, int n_in,
                              void* d_out, int out_size, void* d_ws, size_t ws_size,
                              hipStream_t stream) {
  (void)in_sizes; (void)n_in; (void)out_size; (void)ws_size;
  const float* q_emb = (const float*)d_in[0];
  const float* s_emb = (const float*)d_in[1];
  // d_in[2] = lens (unused in eval mode)
  const float* Wq1 = (const float*)d_in[3];  const float* bq1 = (const float*)d_in[4];
  const float* Wv1 = (const float*)d_in[5];  const float* bv1 = (const float*)d_in[6];
  const float* Wo1 = (const float*)d_in[7];  const float* bo1 = (const float*)d_in[8];
  const float* g1  = (const float*)d_in[9];
  const float* lng1 = (const float*)d_in[10]; const float* lnb1 = (const float*)d_in[11];
  const float* Wq2 = (const float*)d_in[12]; const float* bq2 = (const float*)d_in[13];
  const float* Wv2 = (const float*)d_in[14]; const float* bv2 = (const float*)d_in[15];
  const float* Wo2 = (const float*)d_in[16]; const float* bo2 = (const float*)d_in[17];
  const float* g2  = (const float*)d_in[18];
  const float* lng2 = (const float*)d_in[19]; const float* lnb2 = (const float*)d_in[20];
  const float* Wq3 = (const float*)d_in[21]; const float* bq3 = (const float*)d_in[22];
  const float* Wk3 = (const float*)d_in[23]; const float* bk3 = (const float*)d_in[24];
  const float* Wv3 = (const float*)d_in[25]; const float* bv3 = (const float*)d_in[26];
  const float* Wo3 = (const float*)d_in[27]; const float* bo3 = (const float*)d_in[28];
  const float* g3  = (const float*)d_in[29];
  const float* lng3 = (const float*)d_in[30]; const float* lnb3 = (const float*)d_in[31];
  const float* Wlk = (const float*)d_in[32]; const float* blk = (const float*)d_in[33];
  const float* Wlv = (const float*)d_in[34]; const float* blv = (const float*)d_in[35];
  const float* know = (const float*)d_in[36];

  float* ws = (float*)d_ws;
  const size_t MAT = (size_t)BS * SEQ * DM;   // 2M floats per [B,S,D] buffer
  float* hq   = ws;
  float* hs   = ws + 1 * MAT;
  float* hh   = ws + 2 * MAT;
  float* bufQ = ws + 3 * MAT;
  float* bufV = ws + 4 * MAT;
  float* bufC = ws + 5 * MAT;
  float* bufO = ws + 6 * MAT;
  float* q3   = ws + 7 * MAT;
  float* kkb  = q3 + 256;

  dim3 gGemm(BS * SEQ / 16, 2);
  dim3 gAttn(SEQ / 16, NH, BS);
  dim3 gLN(BS * SEQ / 8);
  dim3 gFin(SEQ, 1, BS);
  const size_t attnSh = 16 * 1024 * 4 + 16 * 1024 * 2;  // 96KB dynamic LDS

  // ---- layer 1 (self-attn on q_emb; K shares Q projection) ----
  gemm256_wt_bias<<<gGemm, 256, 0, stream>>>(q_emb, Wq1, bq1, bufQ);
  gemm256_wt_bias<<<gGemm, 256, 0, stream>>>(q_emb, Wv1, bv1, bufV);
  attn_decay_kernel<<<gAttn, 256, attnSh, stream>>>(bufQ, SEQ * DM, DM, bufQ, bufV, g1, bufC, 1);
  gemm256_wt_bias<<<gGemm, 256, 0, stream>>>(bufC, Wo1, bo1, bufO);
  resln_kernel<<<gLN, 256, 0, stream>>>(q_emb, DM, bufO, lng1, lnb1, hq);

  // ---- layer 2 (self-attn on s_emb) ----
  gemm256_wt_bias<<<gGemm, 256, 0, stream>>>(s_emb, Wq2, bq2, bufQ);
  gemm256_wt_bias<<<gGemm, 256, 0, stream>>>(s_emb, Wv2, bv2, bufV);
  attn_decay_kernel<<<gAttn, 256, attnSh, stream>>>(bufQ, SEQ * DM, DM, bufQ, bufV, g2, bufC, 1);
  gemm256_wt_bias<<<gGemm, 256, 0, stream>>>(bufC, Wo2, bo2, bufO);
  resln_kernel<<<gLN, 256, 0, stream>>>(s_emb, DM, bufO, lng2, lnb2, hs);

  // ---- layer 3 (query = broadcast know; K=proj(hq), V=proj(hs); strict mask)
  know_proj_kernel<<<dim3(1), 256, 0, stream>>>(know, Wq3, bq3, Wlk, blk, q3, kkb);
  gemm256_wt_bias<<<gGemm, 256, 0, stream>>>(hq, Wk3, bk3, bufQ);
  gemm256_wt_bias<<<gGemm, 256, 0, stream>>>(hs, Wv3, bv3, bufV);
  attn_decay_kernel<<<gAttn, 256, attnSh, stream>>>(q3, 0, 0, bufQ, bufV, g3, bufC, 0);
  gemm256_wt_bias<<<gGemm, 256, 0, stream>>>(bufC, Wo3, bo3, bufO);
  resln_kernel<<<gLN, 256, 0, stream>>>(know, 0, bufO, lng3, lnb3, hh);

  // ---- final head mixing ----
  final_mix_kernel<<<gFin, 256, 0, stream>>>(q_emb, hh, kkb, Wlv, blv, (float*)d_out);
}
